// MSC_31671088840802
// MI455X (gfx1250) — matrix operand order repaired and checked by reading, not compiled, not run on hardware
//
#include <hip/hip_runtime.h>
#include <hip/hip_bf16.h>
#include <stdint.h>

typedef __attribute__((ext_vector_type(16))) _Float16 v16h;
typedef __attribute__((ext_vector_type(8)))  _Float16 v8h;
typedef __attribute__((ext_vector_type(8)))  float    v8f;

#define BB    8
#define CC    384
#define NN    1024
#define HEADS 8
#define HD    48
#define HDP   64
#define SCALE 0.14433756729740643f /* 48^-0.5 */

// ---------------------------------------------------------------- helpers

__device__ __forceinline__ v8f wmma_f16(v16h a, v16h b, v8f c) {
  return __builtin_amdgcn_wmma_f32_16x16x32_f16(false, a, false, b,
                                                (short)0, c, false, false);
}

// A-operand: lane group g holds two contiguous 16B chunks: K=[g*16,g*16+8) and
// K=[g*16+8, g*16+16)  (contiguous k order, same as B operand)
union Aop { v16h v; v8h h2[2]; };

// monotonic float -> uint order map
__device__ __forceinline__ unsigned f2o(float f) {
  unsigned u = __float_as_uint(f);
  return (u & 0x80000000u) ? ~u : (u | 0x80000000u);
}

// exact k-th-largest threshold over 1024 row entries (32 per lane) via
// 32-step bitwise bisection; wave-cooperative count with shfl_xor.
__device__ __forceinline__ unsigned topk_thresh(const unsigned* ov, int kk) {
  unsigned t = 0u;
  for (int bit = 31; bit >= 0; --bit) {
    unsigned cand = t | (1u << bit);
    int c = 0;
#pragma unroll
    for (int j = 0; j < 32; ++j) c += (ov[j] >= cand) ? 1 : 0;
    for (int off = 16; off >= 1; off >>= 1) c += __shfl_xor(c, off, 32);
    if (c >= kk) t = cand;
  }
  return t;
}

// ---------------------------------------------------------------- stage 0a: weight f32[K][N] -> f16[N][K]

__global__ __launch_bounds__(256) void wconv_kernel(const float* __restrict__ src,
                                                    _Float16* __restrict__ dst,
                                                    int K, int Ncol) {
  int idx = blockIdx.x * 256 + threadIdx.x;   // dst-contiguous: idx = c*K + k
  if (idx >= K * Ncol) return;
  int k = idx % K, c = idx / K;
  dst[idx] = (_Float16)src[(size_t)k * Ncol + c];
}

// ---------------------------------------------------------------- stage 0b: x[B,C,N] -> xh f16 [B,N,C] (LDS-tiled)

__global__ __launch_bounds__(256) void xconv_kernel(const float* __restrict__ x,
                                                    _Float16* __restrict__ xh) {
  __shared__ float tile[32][33];
  int tx = threadIdx.x & 31, ty = threadIdx.x >> 5;    // 32 x 8
  int n0 = blockIdx.x * 32, c0 = blockIdx.y * 32, b = blockIdx.z;
#pragma unroll
  for (int j = 0; j < 4; ++j) {
    int c = c0 + ty + j * 8;
    tile[ty + j * 8][tx] = x[((size_t)b * CC + c) * NN + n0 + tx];
  }
  __syncthreads();
#pragma unroll
  for (int j = 0; j < 4; ++j) {
    int n = n0 + ty + j * 8;
    xh[((size_t)b * NN + n) * CC + c0 + tx] = (_Float16)tile[tx][ty + j * 8];
  }
}

// ---------------------------------------------------------------- stage 1: pooling

// yp = avgpool3 + avgpool5 + avgpool7 (stride 1, zero pad, count_include_pad)
__global__ __launch_bounds__(256) void pool_kernel(const float* __restrict__ y,
                                                   float* __restrict__ yp) {
  int idx = blockIdx.x * 256 + threadIdx.x;            // over B*C*HW
  if (idx >= BB * CC * NN) return;
  int hw = idx & 1023, wx = hw & 31, hy = hw >> 5;
  const float* src = y + (size_t)(idx >> 10) * NN;
  float acc = 0.f;
  for (int dy = -3; dy <= 3; ++dy) {
    int yy = hy + dy; if (yy < 0 || yy >= 32) continue;
    for (int dx = -3; dx <= 3; ++dx) {
      int xx = wx + dx; if (xx < 0 || xx >= 32) continue;
      int ady = dy < 0 ? -dy : dy, adx = dx < 0 ? -dx : dx;
      float w = 1.f / 49.f;
      if (adx <= 2 && ady <= 2) w += 1.f / 25.f;
      if (adx <= 1 && ady <= 1) w += 1.f / 9.f;
      acc += w * src[yy * 32 + xx];
    }
  }
  yp[idx] = acc;
}

// ---------------------------------------------------------------- stage 2: LayerNorm -> f16 rows

__global__ __launch_bounds__(128) void ln_kernel(const float* __restrict__ yp,
                                                 const float* __restrict__ lnw,
                                                 const float* __restrict__ lnb,
                                                 _Float16* __restrict__ yth) {
  int wave = threadIdx.x >> 5, lane = threadIdx.x & 31;
  int row = blockIdx.x * 4 + wave;                     // 0..8191
  int b = row >> 10, n = row & 1023;
  const float* src = yp + (size_t)b * CC * NN + n;     // stride NN per channel
  float vals[12];
  float s = 0.f;
#pragma unroll
  for (int j = 0; j < 12; ++j) {
    float v = src[(size_t)(lane + 32 * j) * NN];
    vals[j] = v; s += v;
  }
  for (int off = 16; off >= 1; off >>= 1) s += __shfl_xor(s, off, 32);
  float mu = s * (1.f / CC);
  float vv = 0.f;
#pragma unroll
  for (int j = 0; j < 12; ++j) { float d = vals[j] - mu; vv += d * d; }
  for (int off = 16; off >= 1; off >>= 1) vv += __shfl_xor(vv, off, 32);
  float inv = rsqrtf(vv * (1.f / CC) + 1e-5f);
  _Float16* dst = yth + (size_t)row * CC;
#pragma unroll
  for (int j = 0; j < 12; ++j) {
    int c = lane + 32 * j;
    dst[c] = (_Float16)((vals[j] - mu) * inv * lnw[c] + lnb[c]);
  }
}

// ---------------------------------------------------------------- stage 3: Q projection (WMMA, f16-staged)

__global__ __launch_bounds__(32) void qproj_kernel(const _Float16* __restrict__ xh,
                                                   const _Float16* __restrict__ qwt,
                                                   _Float16* __restrict__ qbuf) {
  int lane = threadIdx.x, g = lane >> 4, mr = lane & 15;
  const _Float16* arow = xh + (size_t)(blockIdx.x * 16 + mr) * CC + g * 16;
  const _Float16* bcol = qwt + (size_t)(blockIdx.y * 16 + mr) * CC + g * 16;
  v8f acc = {};
#pragma unroll
  for (int ks = 0; ks < CC / 32; ++ks) {
    Aop A;
    A.h2[0] = *(const v8h*)(arow + ks * 32);
    A.h2[1] = *(const v8h*)(arow + ks * 32 + 8);
    v16h Bv = *(const v16h*)(bcol + ks * 32);
    acc = wmma_f16(A.v, Bv, acc);
  }
  // block-uniform head/d and b/n split (tiles never cross boundaries)
  int head = (blockIdx.y * 16) / HD;
  int d = (blockIdx.y * 16 - head * HD) + mr;
  int bb = (blockIdx.x * 16) >> 10, nb = (blockIdx.x * 16) & 1023;
  _Float16* dst = qbuf + (((size_t)bb * HEADS + head) * NN + nb + 8 * g) * HDP + d;
#pragma unroll
  for (int r = 0; r < 8; ++r) dst[r * HDP] = (_Float16)(acc[r] * SCALE);
}

// ---------------------------------------------------------------- stage 4: KV projection (WMMA, f16-staged)

// col<384 -> K (padded-64 rows) ; col>=384 -> V transposed [bh][d][n]
__global__ __launch_bounds__(32) void kvproj_kernel(const _Float16* __restrict__ yth,
                                                    const _Float16* __restrict__ kvwt,
                                                    _Float16* __restrict__ kbuf,
                                                    _Float16* __restrict__ vt) {
  int lane = threadIdx.x, g = lane >> 4, mr = lane & 15;
  const _Float16* arow = yth + (size_t)(blockIdx.x * 16 + mr) * CC + g * 16;
  const _Float16* bcol = kvwt + (size_t)(blockIdx.y * 16 + mr) * CC + g * 16;
  v8f acc = {};
#pragma unroll
  for (int ks = 0; ks < CC / 32; ++ks) {
    Aop A;
    A.h2[0] = *(const v8h*)(arow + ks * 32);
    A.h2[1] = *(const v8h*)(arow + ks * 32 + 8);
    v16h Bv = *(const v16h*)(bcol + ks * 32);
    acc = wmma_f16(A.v, Bv, acc);
  }
  int bb = (blockIdx.x * 16) >> 10, nb = (blockIdx.x * 16) & 1023;
  if (blockIdx.y < CC / 16) {                          // K half
    int head = (blockIdx.y * 16) / HD;
    int d = (blockIdx.y * 16 - head * HD) + mr;
    _Float16* dst = kbuf + (((size_t)bb * HEADS + head) * NN + nb + 8 * g) * HDP + d;
#pragma unroll
    for (int r = 0; r < 8; ++r) dst[r * HDP] = (_Float16)acc[r];
  } else {                                             // V half -> transposed
    int cb = blockIdx.y * 16 - CC;
    int head = cb / HD;
    int d = (cb - head * HD) + mr;
    _Float16* dst = vt + (((size_t)bb * HEADS + head) * HD + d) * NN + nb + 8 * g;
#pragma unroll
    for (int r = 0; r < 8; ++r) dst[r] = (_Float16)acc[r];
  }
}

// ---------------------------------------------------------------- stage 5: fused top-k attention

// One block = 16 query rows of one (b,h). 4 waves.
//  phase1: QK^T via WMMA into LDS S[16][1024]  (q pre-scaled by 48^-0.5)
//  phase2: per row (4 rows/wave): max, exact top-512 & top-341 thresholds
//          (bitwise bisection on order-mapped floats), exp-sums
//  phase3: PV via WMMA, weights recomputed on the fly from S + row stats
__global__ __launch_bounds__(128) void attn_kernel(const _Float16* __restrict__ qbuf,
                                                   const _Float16* __restrict__ kbuf,
                                                   const _Float16* __restrict__ vt,
                                                   const float* __restrict__ p1p,
                                                   const float* __restrict__ p2p,
                                                   _Float16* __restrict__ Zh) {
  __shared__ float    S[16][NN];        // 64 KB of the 320 KB WGP LDS
  __shared__ float    s_rmax[16], s_c1[16], s_c2[16];
  __shared__ unsigned s_t1[16], s_t2[16];

  int tid = threadIdx.x, wave = tid >> 5, lane = tid & 31;
  int g = lane >> 4, mr = lane & 15;
  int qt = blockIdx.x;
  int bh = blockIdx.z * HEADS + blockIdx.y;

  const _Float16* qbase = qbuf + (size_t)bh * NN * HDP;
  const _Float16* kbase = kbuf + (size_t)bh * NN * HDP;

  // --- phase 1: scores ---------------------------------------------------
  Aop A0, A1;
  {
    const _Float16* qrow = qbase + (size_t)(qt * 16 + mr) * HDP + g * 16;
    A0.h2[0] = *(const v8h*)(qrow);           // d 0..31 (this lane's chunks)
    A0.h2[1] = *(const v8h*)(qrow + 8);
    A1.h2[0] = *(const v8h*)(qrow + 32);      // d 32..63 (zero padded)
    A1.h2[1] = *(const v8h*)(qrow + 40);
  }
  for (int kt = wave * 16; kt < wave * 16 + 16; ++kt) {
    const _Float16* krow = kbase + (size_t)(kt * 16 + mr) * HDP + g * 16;
    v16h B0 = *(const v16h*)(krow);
    v16h B1 = *(const v16h*)(krow + 32);
    v8f acc = {};
    acc = wmma_f16(A0.v, B0, acc);
    acc = wmma_f16(A1.v, B1, acc);
#pragma unroll
    for (int r = 0; r < 8; ++r)
      S[r + 8 * g][kt * 16 + mr] = acc[r];
  }
  __syncthreads();

  // --- phase 2: row stats ------------------------------------------------
  float p1v = p1p[0], p2v = p2p[0];
  for (int rr = 0; rr < 4; ++rr) {
    int r = wave * 4 + rr;
    float sv[32]; unsigned ov[32];
    float mx = -3.4e38f;
#pragma unroll
    for (int j = 0; j < 32; ++j) {
      float s = S[r][lane + 32 * j];
      sv[j] = s; ov[j] = f2o(s);
      mx = fmaxf(mx, s);
    }
    for (int off = 16; off >= 1; off >>= 1) mx = fmaxf(mx, __shfl_xor(mx, off, 32));
    unsigned t1 = topk_thresh(ov, NN / 2);   // k1 = 512
    unsigned t2 = topk_thresh(ov, NN / 3);   // k2 = 341
    float s1 = 0.f, s2 = 0.f;
#pragma unroll
    for (int j = 0; j < 32; ++j) {
      float e = __expf(sv[j] - mx);
      if (ov[j] >= t1) s1 += e;
      if (ov[j] >= t2) s2 += e;
    }
    for (int off = 16; off >= 1; off >>= 1) {
      s1 += __shfl_xor(s1, off, 32);
      s2 += __shfl_xor(s2, off, 32);
    }
    if (lane == 0) {
      s_rmax[r] = mx; s_t1[r] = t1; s_t2[r] = t2;
      s_c1[r] = p1v / s1; s_c2[r] = p2v / s2;
    }
  }
  __syncthreads();

  // --- phase 3: PV -------------------------------------------------------
  if (wave < 3) {                       // wave = d-tile (3 x 16 = 48 dims)
    float rmax = s_rmax[mr];
    unsigned t1 = s_t1[mr], t2 = s_t2[mr];
    float c1 = s_c1[mr], c2 = s_c2[mr];
    const _Float16* vrow = vt + ((size_t)bh * HD + wave * 16 + mr) * NN + g * 16;
    v8f acc = {};
    for (int ks = 0; ks < NN / 32; ++ks) {
      Aop Aw;
#pragma unroll
      for (int i = 0; i < 8; ++i) {
        int kb = ks * 32 + g * 16 + 2 * i;
        float sa = S[mr][kb], sb = S[mr][kb + 1];
        float ea = __expf(sa - rmax), eb = __expf(sb - rmax);
        unsigned oa = f2o(sa), ob = f2o(sb);
        float wa = (oa >= t1 ? c1 * ea : 0.f) + (oa >= t2 ? c2 * ea : 0.f);
        float wb = (ob >= t1 ? c1 * eb : 0.f) + (ob >= t2 ? c2 * eb : 0.f);
        Aw.v[2 * i]     = (_Float16)wa;
        Aw.v[2 * i + 1] = (_Float16)wb;
      }
      v16h Bv = *(const v16h*)(vrow + ks * 32);
      acc = wmma_f16(Aw.v, Bv, acc);
    }
    int col = blockIdx.y * HD + wave * 16 + mr;     // C index = head*48 + d
    _Float16* zdst = Zh + ((size_t)blockIdx.z * NN + qt * 16 + 8 * g) * CC + col;
#pragma unroll
    for (int r = 0; r < 8; ++r) zdst[(size_t)r * CC] = (_Float16)acc[r];
  }
}

// ---------------------------------------------------------------- stage 6: output projection (WMMA)

__global__ __launch_bounds__(32) void outproj_kernel(const _Float16* __restrict__ Zh,
                                                     const _Float16* __restrict__ pwt,
                                                     const float* __restrict__ pb,
                                                     float* __restrict__ out) {
  int lane = threadIdx.x, g = lane >> 4, mr = lane & 15;
  int col = blockIdx.y * 16 + mr;
  const _Float16* arow = Zh + (size_t)(blockIdx.x * 16 + mr) * CC + g * 16;
  const _Float16* bcol = pwt + (size_t)col * CC + g * 16;
  v8f acc = {};
#pragma unroll
  for (int ks = 0; ks < CC / 32; ++ks) {
    Aop A;
    A.h2[0] = *(const v8h*)(arow + ks * 32);
    A.h2[1] = *(const v8h*)(arow + ks * 32 + 8);
    v16h Bv = *(const v16h*)(bcol + ks * 32);
    acc = wmma_f16(A.v, Bv, acc);
  }
  float bias = pb[col];
  int bb = (blockIdx.x * 16) >> 10, nb = (blockIdx.x * 16) & 1023;
  float* odst = out + ((size_t)bb * CC + col) * NN + nb + 8 * g;   // [B,C,H,W]
#pragma unroll
  for (int r = 0; r < 8; ++r) odst[r] = acc[r] + bias;
}

// ---------------------------------------------------------------- launch

extern "C" void kernel_launch(void* const* d_in, const int* in_sizes, int n_in,
                              void* d_out, int out_size, void* d_ws, size_t ws_size,
                              hipStream_t stream) {
  (void)in_sizes; (void)n_in; (void)out_size; (void)ws_size;
  const float* x    = (const float*)d_in[0];
  const float* y    = (const float*)d_in[1];
  const float* q_w  = (const float*)d_in[2];
  const float* kv_w = (const float*)d_in[3];
  const float* ln_w = (const float*)d_in[4];
  const float* ln_b = (const float*)d_in[5];
  const float* pj_w = (const float*)d_in[6];
  const float* pj_b = (const float*)d_in[7];
  const float* p1   = (const float*)d_in[8];
  const float* p2   = (const float*)d_in[9];
  float* out = (float*)d_out;

  char* ws = (char*)d_ws;
  size_t off = 0;
  auto alloc = [&](size_t bytes) -> void* {
    void* p = ws + off;
    off = (off + bytes + 255) & ~(size_t)255;
    return p;
  };
  float*    yp   = (float*)alloc(sizeof(float) * BB * CC * NN);
  _Float16* yth  = (_Float16*)alloc(sizeof(_Float16) * BB * NN * CC);
  _Float16* xh   = (_Float16*)alloc(sizeof(_Float16) * BB * NN * CC);
  _Float16* qwt  = (_Float16*)alloc(sizeof(_Float16) * CC * CC);
  _Float16* kvwt = (_Float16*)alloc(sizeof(_Float16) * CC * 2 * CC);
  _Float16* pwt  = (_Float16*)alloc(sizeof(_Float16) * CC * CC);
  _Float16* qbuf = (_Float16*)alloc(sizeof(_Float16) * BB * HEADS * NN * HDP);
  _Float16* kbuf = (_Float16*)alloc(sizeof(_Float16) * BB * HEADS * NN * HDP);
  _Float16* vt   = (_Float16*)alloc(sizeof(_Float16) * BB * HEADS * HD * NN);
  _Float16* Zh   = (_Float16*)alloc(sizeof(_Float16) * BB * NN * CC);

  // zero the d=48..63 padding of q/k (contiguous allocations)
  hipMemsetAsync(qbuf, 0, sizeof(_Float16) * (size_t)2 * BB * HEADS * NN * HDP, stream);

  wconv_kernel<<<(CC * CC + 255) / 256, 256, 0, stream>>>(q_w, qwt, CC, CC);
  wconv_kernel<<<(CC * 2 * CC + 255) / 256, 256, 0, stream>>>(kv_w, kvwt, CC, 2 * CC);
  wconv_kernel<<<(CC * CC + 255) / 256, 256, 0, stream>>>(pj_w, pwt, CC, CC);
  xconv_kernel<<<dim3(NN / 32, CC / 32, BB), 256, 0, stream>>>(x, xh);
  pool_kernel<<<(BB * CC * NN) / 256, 256, 0, stream>>>(y, yp);
  ln_kernel<<<(BB * NN) / 4, 128, 0, stream>>>(yp, ln_w, ln_b, yth);
  qproj_kernel<<<dim3(BB * NN / 16, CC / 16), 32, 0, stream>>>(xh, qwt, qbuf);
  kvproj_kernel<<<dim3(BB * NN / 16, 2 * CC / 16), 32, 0, stream>>>(yth, kvwt, kbuf, vt);
  attn_kernel<<<dim3(NN / 16, HEADS, BB), 128, 0, stream>>>(qbuf, kbuf, vt, p1, p2, Zh);
  outproj_kernel<<<dim3(BB * NN / 16, CC / 16), 32, 0, stream>>>(Zh, pwt, pj_b, out);
}
